// SS3D_directionGroup_62483184222326
// MI455X (gfx1250) — matrix operand order repaired and checked by reading, not compiled, not run on hardware
//
#include <hip/hip_runtime.h>
#include <hip/hip_bf16.h>
#include <math.h>

typedef __attribute__((ext_vector_type(16))) _Float16 v16h;
typedef __attribute__((ext_vector_type(8)))  _Float16 v8h;
typedef __attribute__((ext_vector_type(8)))  float    v8f;

#define DIMS   24
#define LTOT   13824            // 24^3
#define CDIM   128
#define NSTATE 16
#define TC     216              // chunk length; 2L = 128*216
#define NCH    128              // chunks over the 2L virtual sequence
#define NCHB   64               // backward-phase chunks (produce outputs)

__device__ __forceinline__ float siluf(float s) { return s / (1.f + __expf(-s)); }
__device__ __forceinline__ float softplusf(float s) {
  return (s > 20.f) ? s : log1pf(__expf(s));
}

// Map scan column t (permuted flat index) of direction dir -> spatial flat index
// for the (l = (d*24+w)*24+h, c) layout.
__device__ __forceinline__ int perm_pos(int dir, int t) {
  int t2 = t % DIMS, t1 = (t / DIMS) % DIMS, t0 = t / (DIMS * DIMS);
  int d, w, h;
  switch (dir) {
    case 0: d = t0; h = t1; w = t2; break;   // (D,H,W)
    case 1: h = t0; w = t1; d = t2; break;   // (H,W,D)
    case 2: w = t0; d = t1; h = t2; break;   // (W,D,H)
    default: d = t0; w = t1; h = t2; break;  // (D,W,H)
  }
  return (d * DIMS + w) * DIMS + h;
}

// WMMA f16 16x16x32 fragment loaders (ISA 7.12.2 layouts), operands pre-f16.
// A (16xK slice, row-major, rowstride halfs): half j -> K=(j&7)+16*(j>>3)+8*hi
//   => two contiguous 8-half runs at k0+8*hi and k0+16+8*hi.
__device__ __forceinline__ v16h load_a_frag(const _Float16* p, int hi) {
  union { v16h v; v8h h[2]; } u;
  u.h[0] = *(const v8h*)(p + hi * 8);
  u.h[1] = *(const v8h*)(p + 16 + hi * 8);
  return u.v;
}
// B (Kx16 slice; source stored col-major = W row-major): half j -> K=(j&7)+16*(j>>3)+8*hi
//   => two contiguous 8-half runs at k0+8*hi and k0+16+8*hi.
__device__ __forceinline__ v16h load_b_frag(const _Float16* p, int hi) {
  union { v16h v; v8h h[2]; } u;
  u.h[0] = *(const v8h*)(p + hi * 8);
  u.h[1] = *(const v8h*)(p + 16 + hi * 8);
  return u.v;
}

// ---------------------------------------------------------------------------
// Prep kernels: one-time f32 -> f16 conversions / zero-padded weight panels.
// ---------------------------------------------------------------------------
__global__ __launch_bounds__(256) void k_cvt(const float* __restrict__ src,
                                             _Float16* __restrict__ dst, int n) {
  int i = blockIdx.x * blockDim.x + threadIdx.x;
  if (i < n) dst[i] = (_Float16)src[i];
}
// x_proj_weight (4,40,128) -> padded (4,48,128)
__global__ __launch_bounds__(256) void k_prep_xpw(const float* __restrict__ src,
                                                  _Float16* __restrict__ dst) {
  int i = blockIdx.x * blockDim.x + threadIdx.x;
  if (i >= 4 * 48 * 128) return;
  int k = i & 127, col = (i >> 7) % 48, dir = i / (48 * 128);
  dst[i] = (col < 40) ? (_Float16)src[(dir * 40 + col) * 128 + k] : (_Float16)0.f;
}
// dt_projs_weight (4,128,8) -> padded (4,128,32)
__global__ __launch_bounds__(256) void k_prep_dtw(const float* __restrict__ src,
                                                  _Float16* __restrict__ dst) {
  int i = blockIdx.x * blockDim.x + threadIdx.x;
  if (i >= 4 * 128 * 32) return;
  int k = i & 31, n = (i >> 5) & 127, dir = i / (128 * 32);
  dst[i] = (k < 8) ? (_Float16)src[(dir * 128 + n) * 8 + k] : (_Float16)0.f;
}
__global__ __launch_bounds__(256) void k_zero16(_Float16* __restrict__ p, int n) {
  int i = blockIdx.x * blockDim.x + threadIdx.x;
  if (i < n) p[i] = (_Float16)0.f;
}

// ---------------------------------------------------------------------------
// GEMM 1: xz[L x 256] = x[L x 128] * in_proj_w^T. Wave = 16 rows x 64 cols.
// ---------------------------------------------------------------------------
__global__ __launch_bounds__(256) void k_inproj(const _Float16* __restrict__ x16,
                                                const _Float16* __restrict__ w16,
                                                float* __restrict__ xz) {
  int wid = blockIdx.x * (blockDim.x >> 5) + (threadIdx.x >> 5);
  if (wid >= (LTOT / 16) * 4) return;
  int tm = wid >> 2, tg = wid & 3;
  int lane = threadIdx.x & 31, r = lane & 15, hi = lane >> 4;
  const _Float16* arow = x16 + (tm * 16 + r) * 128;
  v8f acc[4] = {};
#pragma unroll
  for (int k0 = 0; k0 < 128; k0 += 32) {
    v16h a = load_a_frag(arow + k0, hi);
#pragma unroll
    for (int nt = 0; nt < 4; ++nt) {
      int col = (tg * 4 + nt) * 16 + r;
      v16h b = load_b_frag(w16 + col * 128 + k0, hi);
      acc[nt] = __builtin_amdgcn_wmma_f32_16x16x32_f16(false, a, false, b,
                                                       (short)0, acc[nt],
                                                       false, false);
    }
  }
#pragma unroll
  for (int nt = 0; nt < 4; ++nt)
#pragma unroll
    for (int v = 0; v < 8; ++v)
      xz[(tm * 16 + v + hi * 8) * 256 + (tg * 4 + nt) * 16 + r] = acc[nt][v];
}

// ---------------------------------------------------------------------------
// Depthwise 3x3x3 conv (pad 1) + bias + SiLU; writes f32 and f16 mirrors.
// ---------------------------------------------------------------------------
__global__ __launch_bounds__(256) void k_conv(const float* __restrict__ xz,
                                              const float* __restrict__ cw,
                                              const float* __restrict__ cb,
                                              float* __restrict__ xc,
                                              _Float16* __restrict__ xc16) {
  int idx = blockIdx.x * blockDim.x + threadIdx.x;
  if (idx >= LTOT * CDIM) return;
  int c = idx & 127, l = idx >> 7;
  int h = l % DIMS, w = (l / DIMS) % DIMS, d = l / (DIMS * DIMS);
  float s = cb[c];
#pragma unroll
  for (int dz = -1; dz <= 1; ++dz) {
    int zd = d + dz;
    if ((unsigned)zd >= DIMS) continue;
#pragma unroll
    for (int dy = -1; dy <= 1; ++dy) {
      int yw = w + dy;
      if ((unsigned)yw >= DIMS) continue;
#pragma unroll
      for (int dx = -1; dx <= 1; ++dx) {
        int xh = h + dx;
        if ((unsigned)xh >= DIMS) continue;
        float wv = cw[c * 27 + (dz + 1) * 9 + (dy + 1) * 3 + (dx + 1)];
        s += wv * xz[((zd * DIMS + yw) * DIMS + xh) * 256 + c];
      }
    }
  }
  float o = siluf(s);
  xc[l * CDIM + c] = o;
  xc16[l * CDIM + c] = (_Float16)o;
}

// ---------------------------------------------------------------------------
// GEMM 2 (per dir): x_dbl[L x 48pad] = gather(xc16, perm) * xpw16[dir]^T.
// One wave covers all 3 col-tiles (A loaded once). cols 0..7 -> dtr16 (f16,
// L x 32 zero-padded panel), 8..23 -> Bs, 24..39 -> Cs.
// ---------------------------------------------------------------------------
__global__ __launch_bounds__(256) void k_xdbl(const _Float16* __restrict__ xc16,
                                              const _Float16* __restrict__ xpw16,
                                              int dir,
                                              _Float16* __restrict__ dtr16,
                                              float* __restrict__ Bs,
                                              float* __restrict__ Cs) {
  int tm = blockIdx.x * (blockDim.x >> 5) + (threadIdx.x >> 5);
  if (tm >= LTOT / 16) return;
  int lane = threadIdx.x & 31, r = lane & 15, hi = lane >> 4;
  const _Float16* W = xpw16 + dir * 48 * 128;
  const _Float16* arow = xc16 + perm_pos(dir, tm * 16 + r) * CDIM;
  v8f acc[3] = {};
#pragma unroll
  for (int k0 = 0; k0 < 128; k0 += 32) {
    v16h a = load_a_frag(arow + k0, hi);
#pragma unroll
    for (int nt = 0; nt < 3; ++nt) {
      int col = nt * 16 + r;
      v16h b = load_b_frag(W + col * 128 + k0, hi);
      acc[nt] = __builtin_amdgcn_wmma_f32_16x16x32_f16(false, a, false, b,
                                                       (short)0, acc[nt],
                                                       false, false);
    }
  }
#pragma unroll
  for (int nt = 0; nt < 3; ++nt)
#pragma unroll
    for (int v = 0; v < 8; ++v) {
      int m = tm * 16 + v + hi * 8;
      int n = nt * 16 + r;
      float val = acc[nt][v];
      if (n < 8)       dtr16[m * 32 + n] = (_Float16)val;
      else if (n < 24) Bs[m * 16 + (n - 8)] = val;
      else if (n < 40) Cs[m * 16 + (n - 24)] = val;
    }
}

// ---------------------------------------------------------------------------
// GEMM 3 (per dir): dts[L x 128] = dtr16[L x 32pad] * dtw16[dir]^T (K=32).
// Wave = 16 rows x 64 cols (4 tiles, one WMMA each).
// ---------------------------------------------------------------------------
__global__ __launch_bounds__(256) void k_dtproj(const _Float16* __restrict__ dtr16,
                                                const _Float16* __restrict__ dtw16,
                                                int dir, float* __restrict__ dts) {
  int wid = blockIdx.x * (blockDim.x >> 5) + (threadIdx.x >> 5);
  if (wid >= (LTOT / 16) * 2) return;
  int tm = wid >> 1, tg = wid & 1;
  int lane = threadIdx.x & 31, r = lane & 15, hi = lane >> 4;
  const _Float16* W = dtw16 + dir * 128 * 32;
  v16h a = load_a_frag(dtr16 + (tm * 16 + r) * 32, hi);
  v8f acc[4] = {};
#pragma unroll
  for (int nt = 0; nt < 4; ++nt) {
    int col = (tg * 4 + nt) * 16 + r;
    v16h b = load_b_frag(W + col * 32, hi);
    acc[nt] = __builtin_amdgcn_wmma_f32_16x16x32_f16(false, a, false, b,
                                                     (short)0, acc[nt],
                                                     false, false);
  }
#pragma unroll
  for (int nt = 0; nt < 4; ++nt)
#pragma unroll
    for (int v = 0; v < 8; ++v)
      dts[(tm * 16 + v + hi * 8) * 128 + (tg * 4 + nt) * 16 + r] = acc[nt][v];
}

// ---------------------------------------------------------------------------
// Scan pass A: per (chunk, c, n) compute chunk transition (P = prod a, Q).
// Virtual 2L sequence: s<L forward column s, s>=L column 2L-1-s.
// ---------------------------------------------------------------------------
__global__ __launch_bounds__(256) void k_chunkA(
    const float* __restrict__ xc, const float* __restrict__ dts,
    const float* __restrict__ Bs, const float* __restrict__ A_logs,
    const float* __restrict__ dtb, int dir, float* __restrict__ P,
    float* __restrict__ Q) {
  int idx = blockIdx.x * blockDim.x + threadIdx.x;
  if (idx >= NCH * CDIM * NSTATE) return;
  int n = idx & 15, c = (idx >> 4) & 127, ch = idx >> 11;
  float A = -__expf(A_logs[(dir * CDIM + c) * NSTATE + n]);
  float bias = dtb[dir * CDIM + c];
  float p = 1.f, q = 0.f;
  for (int s = ch * TC; s < (ch + 1) * TC; ++s) {
    int t = (s < LTOT) ? s : (2 * LTOT - 1 - s);
    float dtv = softplusf(dts[t * CDIM + c] + bias);
    float a = __expf(dtv * A);
    float xv = xc[perm_pos(dir, t) * CDIM + c];
    p *= a;
    q = a * q + dtv * xv * Bs[t * NSTATE + n];
  }
  P[idx] = p;
  Q[idx] = q;
}

// Scan pass B: sequential combine over 128 chunks (per c,n lane).
__global__ __launch_bounds__(256) void k_chunkB(const float* __restrict__ P,
                                                const float* __restrict__ Q,
                                                float* __restrict__ Hs) {
  int idx = blockIdx.x * blockDim.x + threadIdx.x;
  if (idx >= CDIM * NSTATE) return;
  float h = 0.f;
  for (int ch = 0; ch < NCH; ++ch) {
    Hs[ch * (CDIM * NSTATE) + idx] = h;
    h = P[ch * (CDIM * NSTATE) + idx] * h + Q[ch * (CDIM * NSTATE) + idx];
  }
}

// Scan pass C: replay backward-phase chunks (s>=L) from Hs, emit outputs.
// 16 lanes (states) per (c, chunk); h*C reduced with shfl_xor.
// Output for step s lands at permuted position (s - L).
__global__ __launch_bounds__(256) void k_chunkC(
    const float* __restrict__ xc, const float* __restrict__ dts,
    const float* __restrict__ Bs, const float* __restrict__ Cs,
    const float* __restrict__ A_logs, const float* __restrict__ Ds,
    const float* __restrict__ dtb, const float* __restrict__ Hs, int dir,
    int accum, float* __restrict__ ysum) {
  int gid = (blockIdx.x * blockDim.x + threadIdx.x) >> 4;
  int n = threadIdx.x & 15;
  if (gid >= NCHB * CDIM) return;
  int c = gid & 127;
  int ch = (gid >> 7) + (NCH - NCHB);  // chunks 64..127
  float A = -__expf(A_logs[(dir * CDIM + c) * NSTATE + n]);
  float bias = dtb[dir * CDIM + c];
  float Dv = Ds[dir * CDIM + c];
  float h = Hs[ch * (CDIM * NSTATE) + c * NSTATE + n];
  for (int s = ch * TC; s < (ch + 1) * TC; ++s) {
    int t = 2 * LTOT - 1 - s;
    float dtv = softplusf(dts[t * CDIM + c] + bias);
    float a = __expf(dtv * A);
    float xv = xc[perm_pos(dir, t) * CDIM + c];
    h = a * h + dtv * xv * Bs[t * NSTATE + n];
    float yp = h * Cs[t * NSTATE + n];
#pragma unroll
    for (int off = 8; off; off >>= 1) yp += __shfl_xor(yp, off);
    if (n == 0) {
      int pos = perm_pos(dir, s - LTOT);
      float y = yp + Dv * xv;
      float* dst = &ysum[pos * CDIM + c];
      *dst = accum ? (*dst + y) : y;
    }
  }
}

// ---------------------------------------------------------------------------
// LayerNorm over C=128 + SiLU(z) gate. One wave per row; emits f16 for GEMM 4.
// ---------------------------------------------------------------------------
__global__ __launch_bounds__(256) void k_lngate(const float* __restrict__ ysum,
                                                const float* __restrict__ xz,
                                                const float* __restrict__ lnw,
                                                const float* __restrict__ lnb,
                                                _Float16* __restrict__ yg16) {
  int row = blockIdx.x * (blockDim.x >> 5) + (threadIdx.x >> 5);
  if (row >= LTOT) return;
  int lane = threadIdx.x & 31;
  float v[4], s = 0.f;
#pragma unroll
  for (int j = 0; j < 4; ++j) {
    v[j] = ysum[row * CDIM + lane * 4 + j];
    s += v[j];
  }
#pragma unroll
  for (int off = 16; off; off >>= 1) s += __shfl_xor(s, off);
  float mu = s * (1.f / 128.f);
  float var = 0.f;
#pragma unroll
  for (int j = 0; j < 4; ++j) {
    float d = v[j] - mu;
    var += d * d;
  }
#pragma unroll
  for (int off = 16; off; off >>= 1) var += __shfl_xor(var, off);
  float rstd = rsqrtf(var * (1.f / 128.f) + 1e-5f);
#pragma unroll
  for (int j = 0; j < 4; ++j) {
    int c = lane * 4 + j;
    float z = xz[row * 256 + 128 + c];
    yg16[row * CDIM + c] =
        (_Float16)(((v[j] - mu) * rstd * lnw[c] + lnb[c]) * siluf(z));
  }
}

// ---------------------------------------------------------------------------
// GEMM 4: out[L x 128] = yg16[L x 128] * out_proj_w^T. Wave = 16 x 64.
// ---------------------------------------------------------------------------
__global__ __launch_bounds__(256) void k_outproj(const _Float16* __restrict__ yg16,
                                                 const _Float16* __restrict__ w16,
                                                 float* __restrict__ out) {
  int wid = blockIdx.x * (blockDim.x >> 5) + (threadIdx.x >> 5);
  if (wid >= (LTOT / 16) * 2) return;
  int tm = wid >> 1, tg = wid & 1;
  int lane = threadIdx.x & 31, r = lane & 15, hi = lane >> 4;
  const _Float16* arow = yg16 + (tm * 16 + r) * 128;
  v8f acc[4] = {};
#pragma unroll
  for (int k0 = 0; k0 < 128; k0 += 32) {
    v16h a = load_a_frag(arow + k0, hi);
#pragma unroll
    for (int nt = 0; nt < 4; ++nt) {
      int col = (tg * 4 + nt) * 16 + r;
      v16h b = load_b_frag(w16 + col * 128 + k0, hi);
      acc[nt] = __builtin_amdgcn_wmma_f32_16x16x32_f16(false, a, false, b,
                                                       (short)0, acc[nt],
                                                       false, false);
    }
  }
#pragma unroll
  for (int nt = 0; nt < 4; ++nt)
#pragma unroll
    for (int v = 0; v < 8; ++v)
      out[(tm * 16 + v + hi * 8) * 128 + (tg * 4 + nt) * 16 + r] = acc[nt][v];
}

extern "C" void kernel_launch(void* const* d_in, const int* in_sizes, int n_in,
                              void* d_out, int out_size, void* d_ws,
                              size_t ws_size, hipStream_t stream) {
  (void)in_sizes; (void)n_in; (void)out_size; (void)ws_size;
  const float* x          = (const float*)d_in[0];
  const float* in_proj_w  = (const float*)d_in[1];
  const float* conv_w     = (const float*)d_in[2];
  const float* conv_b     = (const float*)d_in[3];
  const float* x_proj_w   = (const float*)d_in[4];
  const float* dt_projs_w = (const float*)d_in[5];
  const float* dt_projs_b = (const float*)d_in[6];
  const float* A_logs     = (const float*)d_in[7];
  const float* Ds         = (const float*)d_in[8];
  const float* ln_w       = (const float*)d_in[9];
  const float* ln_b       = (const float*)d_in[10];
  const float* out_proj_w = (const float*)d_in[11];
  float* out = (float*)d_out;

  char* cur = (char*)d_ws;
  auto allocF = [&](size_t n) { float* p = (float*)cur; cur += n * 4; return p; };
  auto allocH = [&](size_t n) { _Float16* p = (_Float16*)cur; cur += n * 2; return p; };

  float* xz   = allocF((size_t)LTOT * 256);
  float* xc   = allocF((size_t)LTOT * CDIM);
  float* Bsb  = allocF((size_t)LTOT * NSTATE);
  float* Csb  = allocF((size_t)LTOT * NSTATE);
  float* dts  = allocF((size_t)LTOT * CDIM);
  float* P    = allocF((size_t)NCH * CDIM * NSTATE);
  float* Q    = allocF((size_t)NCH * CDIM * NSTATE);
  float* Hs   = allocF((size_t)NCH * CDIM * NSTATE);
  float* ysum = allocF((size_t)LTOT * CDIM);
  _Float16* x16   = allocH((size_t)LTOT * CDIM);
  _Float16* xc16  = allocH((size_t)LTOT * CDIM);
  _Float16* yg16  = allocH((size_t)LTOT * CDIM);
  _Float16* dtr16 = allocH((size_t)LTOT * 32);
  _Float16* w16i  = allocH(256 * 128);
  _Float16* w16o  = allocH(128 * 128);
  _Float16* xpw16 = allocH(4 * 48 * 128);
  _Float16* dtw16 = allocH(4 * 128 * 32);

  // --- one-time operand prep (f16 panels) ---
  k_cvt<<<(LTOT * CDIM + 255) / 256, 256, 0, stream>>>(x, x16, LTOT * CDIM);
  k_cvt<<<(256 * 128 + 255) / 256, 256, 0, stream>>>(in_proj_w, w16i, 256 * 128);
  k_cvt<<<(128 * 128 + 255) / 256, 256, 0, stream>>>(out_proj_w, w16o, 128 * 128);
  k_prep_xpw<<<(4 * 48 * 128 + 255) / 256, 256, 0, stream>>>(x_proj_w, xpw16);
  k_prep_dtw<<<(4 * 128 * 32 + 255) / 256, 256, 0, stream>>>(dt_projs_w, dtw16);
  k_zero16<<<(LTOT * 32 + 255) / 256, 256, 0, stream>>>(dtr16, LTOT * 32);

  // 1) in_proj: 864 row-tiles x 4 col-groups = 3456 waves, 8 waves/block
  k_inproj<<<3456 / 8, 256, 0, stream>>>(x16, w16i, xz);
  // 2) depthwise conv + silu (f32 + f16 mirror)
  k_conv<<<(LTOT * CDIM + 255) / 256, 256, 0, stream>>>(xz, conv_w, conv_b, xc,
                                                        xc16);
  // 3) four directions, scratch reused stream-ordered
  for (int dir = 0; dir < 4; ++dir) {
    k_xdbl<<<864 / 8, 256, 0, stream>>>(xc16, xpw16, dir, dtr16, Bsb, Csb);
    k_dtproj<<<1728 / 8, 256, 0, stream>>>(dtr16, dtw16, dir, dts);
    k_chunkA<<<(NCH * CDIM * NSTATE) / 256, 256, 0, stream>>>(
        xc, dts, Bsb, A_logs, dt_projs_b, dir, P, Q);
    k_chunkB<<<(CDIM * NSTATE) / 256, 256, 0, stream>>>(P, Q, Hs);
    k_chunkC<<<(NCHB * CDIM * 16) / 256, 256, 0, stream>>>(
        xc, dts, Bsb, Csb, A_logs, Ds, dt_projs_b, Hs, dir, dir == 0 ? 0 : 1,
        ysum);
  }
  // 4) LN + gate (emits f16), then out_proj
  k_lngate<<<LTOT / 8, 256, 0, stream>>>(ysum, xz, ln_w, ln_b, yg16);
  k_outproj<<<1728 / 8, 256, 0, stream>>>(yg16, w16o, out);
}
